// BlockTransformerMixer_1683627180229
// MI455X (gfx1250) — matrix operand order repaired and checked by reading, not compiled, not run on hardware
//
#include <hip/hip_runtime.h>

// ---------------------------------------------------------------------------
// CDNA5 (gfx1250) block-transformer mixer.
// B=8, T=8192, D=256, H=4, hd=64, L=2, BS=16  ->  4096 independent 16x256
// blocks.  All GEMMs use V_WMMA_F32_16X16X32_BF16 (f32 accumulation).
// ---------------------------------------------------------------------------

typedef __bf16 bf16_t;
typedef __attribute__((ext_vector_type(16))) __bf16 v16bf;
typedef __attribute__((ext_vector_type(8)))  __bf16 v8bf;
typedef __attribute__((ext_vector_type(8)))  float  v8f;

#define TD   256    // model dim
#define TBS  16     // tokens per block
#define THD  64     // head dim
#define TNH  4      // heads
#define NTHREADS 128

// padded LDS row strides (bf16 elems) to spread the 64 LDS banks
#define LDA 264     // [16][256] activation tiles (row = 528B)
#define LDM 520     // [16][512] ff half buffer   (row = 1040B)
#define LDV 24      // vt [256][16] (row = 48B)
#define LDT 24      // attn [16][16] rows (48B)

__device__ __forceinline__ v8bf zero8() {
  v8bf z;
#pragma unroll
  for (int i = 0; i < 8; ++i) z[i] = (__bf16)0.0f;
  return z;
}

__device__ __forceinline__ v16bf cat8(v8bf lo, v8bf hi) {
  return __builtin_shufflevector(lo, hi, 0,1,2,3,4,5,6,7,8,9,10,11,12,13,14,15);
}

// A fragment (16x32, M=row, K contiguous) from a row-major bf16 matrix.
// base points at [row 0][k0]; lane l<16 holds K {k0..k0+7, k0+16..k0+23},
// lane l>=16 holds K {k0+8..k0+15, k0+24..k0+31}  (ISA 7.12.2 16-bit A).
__device__ __forceinline__ v16bf frag_a(const bf16_t* base, int ld, int lane) {
  const bf16_t* p = base + (lane & 15) * ld + ((lane >> 4) << 3);
  v8bf lo = *(const v8bf*)p;
  v8bf hi = *(const v8bf*)(p + 16);
  return cat8(lo, hi);
}

// B fragment (32x16, K x N) where B[k][n] = w[colbase+n][k] (w row-major,
// i.e. x @ w^T).  Lane l holds column (l&15), K = k0 + (l>>4)*8 .. +7 and
// k0 + 16 + (l>>4)*8 .. +7 (same K order as frag_a) -> two 16B loads.
__device__ __forceinline__ v16bf frag_b(const bf16_t* w, int ld, int colbase,
                                        int k0, int lane) {
  const bf16_t* p = w + (size_t)(colbase + (lane & 15)) * ld + k0 + ((lane >> 4) << 3);
  v8bf lo = *(const v8bf*)p;
  v8bf hi = *(const v8bf*)(p + 16);
  return cat8(lo, hi);
}

__device__ __forceinline__ v8f wmma_bf(v16bf a, v16bf b, v8f c) {
  return __builtin_amdgcn_wmma_f32_16x16x32_bf16(false, a, false, b,
                                                 (short)0, c, false, false);
}

// RMS norm of xs[16][256] -> bf16 abf[16][264], 128 threads.
__device__ __forceinline__ void rmsnorm128(const float (*xs)[TD],
                                           bf16_t (*abf)[LDA],
                                           float* red, float* rrms,
                                           const float* nw, int tid) {
  const int row = tid >> 3, sub = tid & 7;
  float s = 0.f;
#pragma unroll
  for (int j = 0; j < 32; ++j) { float v = xs[row][sub * 32 + j]; s += v * v; }
  red[tid] = s;
  __syncthreads();
  if (tid < TBS) {
    float t = 0.f;
#pragma unroll
    for (int j = 0; j < 8; ++j) t += red[tid * 8 + j];
    rrms[tid] = rsqrtf(t * (1.0f / TD) + 1e-6f);
  }
  __syncthreads();
  for (int i = tid; i < TBS * TD; i += NTHREADS) {
    const int m = i >> 8, n = i & 255;
    abf[m][n] = (bf16_t)(xs[m][n] * rrms[m] * nw[n]);
  }
  __syncthreads();
}

__global__ void cvt_f32_bf16(const float* __restrict__ s, bf16_t* __restrict__ d,
                             int n) {
  int i = blockIdx.x * blockDim.x + threadIdx.x;
  if (i < n) d[i] = (bf16_t)s[i];
}

__global__ __launch_bounds__(NTHREADS) void block_tf(
    const float* __restrict__ h,
    const float* __restrict__ n1w, const float* __restrict__ ipb,
    const float* __restrict__ opb, const float* __restrict__ n2w,
    const float* __restrict__ f1b, const float* __restrict__ f2b,
    const bf16_t* __restrict__ ipw, const bf16_t* __restrict__ opw,
    const bf16_t* __restrict__ f1w, const bf16_t* __restrict__ f2w,
    float* __restrict__ out, int L) {
  __shared__ float  xs[TBS][TD];        // residual stream (f32)      16384 B
  __shared__ bf16_t abf[TBS][LDA];      // normed acts / attn out      8448 B
  __shared__ __align__(32) char un[36352];  // phase-union scratch
  __shared__ float  red[NTHREADS];
  __shared__ float  rrms[TBS];

  // attention-phase views of `un`
  bf16_t (*qbf)[LDA] = (bf16_t(*)[LDA])(un);            //  8448 B
  bf16_t (*kbf)[LDA] = (bf16_t(*)[LDA])(un + 8448);     //  8448 B
  bf16_t (*vt)[LDV]  = (bf16_t(*)[LDV])(un + 16896);    // v^T [256][16] 12288 B
  float  (*scores)[TBS][TBS]  = (float(*)[TBS][TBS])(un + 29184);   // 4096 B
  bf16_t (*attnb)[TBS][LDT]   = (bf16_t(*)[TBS][LDT])(un + 33280);  // 3072 B
  // ff-phase view (overlaps q/k only; lifetimes disjoint)
  bf16_t (*mbf)[LDM] = (bf16_t(*)[LDM])(un);            // 16640 B

  const int tid  = threadIdx.x;
  const int lane = tid & 31;
  const int wave = tid >> 5;
  const size_t blk = blockIdx.x;

  // ---- load x block (16x256 f32) -----------------------------------------
  {
    const float4* hp = (const float4*)(h + blk * (TBS * TD));
    float4* xp = (float4*)(&xs[0][0]);
#pragma unroll
    for (int i = 0; i < (TBS * TD / 4) / NTHREADS; ++i)
      xp[tid + i * NTHREADS] = hp[tid + i * NTHREADS];
  }
  __syncthreads();

  for (int l = 0; l < L; ++l) {
    const float*  n1_l  = n1w + (size_t)l * TD;
    const float*  ipb_l = ipb + (size_t)l * 3 * TD;
    const float*  opb_l = opb + (size_t)l * TD;
    const float*  n2_l  = n2w + (size_t)l * TD;
    const float*  f1b_l = f1b + (size_t)l * 4 * TD;
    const float*  f2b_l = f2b + (size_t)l * TD;
    const bf16_t* ipw_l = ipw + (size_t)l * 3 * TD * TD;
    const bf16_t* opw_l = opw + (size_t)l * TD * TD;
    const bf16_t* f1w_l = f1w + (size_t)l * 4 * TD * TD;
    const bf16_t* f2w_l = f2w + (size_t)l * TD * 4 * TD;

    // ---- norm1 -> abf (bf16) --------------------------------------------
    rmsnorm128(xs, abf, red, rrms, n1_l, tid);

    // ---- QKV: [16,256] x [768,256]^T, 48 N-tiles over 4 waves -----------
    // Split into Q / K / V tile ranges so each range has a single,
    // divergence-free store path (q/k row-major, v transposed).
    {
      v16bf av[8];
#pragma unroll
      for (int kk = 0; kk < 8; ++kk) av[kk] = frag_a(&abf[0][kk * 32], LDA, lane);
      const int c = lane & 15, mb = (lane >> 4) * 8;

      // Q tiles: cols [0,256)
      for (int t = wave; t < 16; t += 4) {
        const int nb = t * 16;
        const float bias = ipb_l[nb + c];
        v8f acc;
#pragma unroll
        for (int i = 0; i < 8; ++i) acc[i] = bias;
#pragma unroll
        for (int kk = 0; kk < 8; ++kk)
          acc = wmma_bf(av[kk], frag_b(ipw_l, TD, nb, kk * 32, lane), acc);
#pragma unroll
        for (int i = 0; i < 8; ++i) qbf[mb + i][nb + c] = (bf16_t)acc[i];
      }
      // K tiles: cols [256,512)
      for (int t = 16 + wave; t < 32; t += 4) {
        const int nb = t * 16;
        const float bias = ipb_l[nb + c];
        v8f acc;
#pragma unroll
        for (int i = 0; i < 8; ++i) acc[i] = bias;
#pragma unroll
        for (int kk = 0; kk < 8; ++kk)
          acc = wmma_bf(av[kk], frag_b(ipw_l, TD, nb, kk * 32, lane), acc);
#pragma unroll
        for (int i = 0; i < 8; ++i) kbf[mb + i][nb - 256 + c] = (bf16_t)acc[i];
      }
      // V tiles: cols [512,768), stored transposed into vt[dim][token]
      for (int t = 32 + wave; t < 48; t += 4) {
        const int nb = t * 16;
        const float bias = ipb_l[nb + c];
        v8f acc;
#pragma unroll
        for (int i = 0; i < 8; ++i) acc[i] = bias;
#pragma unroll
        for (int kk = 0; kk < 8; ++kk)
          acc = wmma_bf(av[kk], frag_b(ipw_l, TD, nb, kk * 32, lane), acc);
#pragma unroll
        for (int i = 0; i < 8; ++i) vt[nb - 512 + c][mb + i] = (bf16_t)acc[i];
      }
    }
    __syncthreads();

    // ---- scores = q k^T / 8 : one head per wave, K=64 -------------------
    {
      const int hh = wave;
      v8f acc;
#pragma unroll
      for (int i = 0; i < 8; ++i) acc[i] = 0.f;
#pragma unroll
      for (int kk = 0; kk < 2; ++kk) {
        v16bf a = frag_a(&qbf[0][hh * THD + kk * 32], LDA, lane);
        v16bf b = frag_b(&kbf[0][0], LDA, 0, hh * THD + kk * 32, lane);
        acc = wmma_bf(a, b, acc);
      }
      const int c = lane & 15, mb = (lane >> 4) * 8;
#pragma unroll
      for (int i = 0; i < 8; ++i) scores[hh][mb + i][c] = acc[i] * 0.125f;
    }
    __syncthreads();

    // ---- softmax (f32), one row per thread ------------------------------
    if (tid < TNH * TBS) {
      const int hh = tid >> 4, r = tid & 15;
      const float* s = scores[hh][r];
      float mx = s[0];
#pragma unroll
      for (int j = 1; j < 16; ++j) mx = fmaxf(mx, s[j]);
      float e[16]; float sum = 0.f;
#pragma unroll
      for (int j = 0; j < 16; ++j) { e[j] = expf(s[j] - mx); sum += e[j]; }
      const float inv = 1.0f / sum;
#pragma unroll
      for (int j = 0; j < 16; ++j) attnb[hh][r][j] = (bf16_t)(e[j] * inv);
    }
    __syncthreads();

    // ---- o = attn x v : K=16 zero-padded to 32 --------------------------
    {
      const int hh = wave;
      const int r = lane & 15, koff = (lane >> 4) * 8;
      v8bf alo = *(const v8bf*)&attnb[hh][r][koff];
      v16bf a = cat8(alo, zero8());
#pragma unroll
      for (int t = 0; t < 4; ++t) {
        const int nb = t * 16;
        v16bf b;
        {
          v8bf lo = *(const v8bf*)&vt[hh * THD + nb + r][koff];
          b = cat8(lo, zero8());   // K=16..31 padding
        }
        v8f acc;
#pragma unroll
        for (int i = 0; i < 8; ++i) acc[i] = 0.f;
        acc = wmma_bf(a, b, acc);
        const int c = lane & 15, mb = (lane >> 4) * 8;
#pragma unroll
        for (int i = 0; i < 8; ++i)
          abf[mb + i][hh * THD + nb + c] = (bf16_t)acc[i];   // o into abf
      }
    }
    __syncthreads();

    // ---- out proj + residual: xs += o x opw^T ---------------------------
    {
      v16bf av[8];
#pragma unroll
      for (int kk = 0; kk < 8; ++kk) av[kk] = frag_a(&abf[0][kk * 32], LDA, lane);
      for (int t = wave; t < 16; t += 4) {
        const int nb = t * 16;
        const float bias = opb_l[nb + (lane & 15)];
        v8f acc;
#pragma unroll
        for (int i = 0; i < 8; ++i) acc[i] = bias;
#pragma unroll
        for (int kk = 0; kk < 8; ++kk)
          acc = wmma_bf(av[kk], frag_b(opw_l, TD, nb, kk * 32, lane), acc);
        const int c = lane & 15, mb = (lane >> 4) * 8;
#pragma unroll
        for (int i = 0; i < 8; ++i) xs[mb + i][nb + c] += acc[i];
      }
    }
    __syncthreads();

    // ---- norm2 -> abf ----------------------------------------------------
    rmsnorm128(xs, abf, red, rrms, n2_l, tid);

    // ---- FF: ff1 (gelu) in two 512-wide halves; ff2 accumulates in regs -
    v8f facc[4];
#pragma unroll
    for (int t = 0; t < 4; ++t) {
      const float bias = f2b_l[(wave * 4 + t) * 16 + (lane & 15)];
#pragma unroll
      for (int i = 0; i < 8; ++i) facc[t][i] = bias;
    }
    for (int half = 0; half < 2; ++half) {
      {
        v16bf av[8];
#pragma unroll
        for (int kk = 0; kk < 8; ++kk) av[kk] = frag_a(&abf[0][kk * 32], LDA, lane);
        for (int t = wave; t < 32; t += 4) {
          const int nb = half * 512 + t * 16;
          const float bias = f1b_l[nb + (lane & 15)];
          v8f acc;
#pragma unroll
          for (int i = 0; i < 8; ++i) acc[i] = bias;
#pragma unroll
          for (int kk = 0; kk < 8; ++kk)
            acc = wmma_bf(av[kk], frag_b(f1w_l, TD, nb, kk * 32, lane), acc);
          const int c = lane & 15, mb = (lane >> 4) * 8;
#pragma unroll
          for (int i = 0; i < 8; ++i) {
            const float v = acc[i];
            const float g = 0.5f * v * (1.0f + erff(v * 0.70710678118f)); // exact gelu
            mbf[mb + i][t * 16 + c] = (bf16_t)g;
          }
        }
      }
      __syncthreads();
      // ff2 partial: K = 512 this half; A reused across the wave's 4 tiles
      for (int k0 = 0; k0 < 512; k0 += 32) {
        v16bf a = frag_a(&mbf[0][k0], LDM, lane);
#pragma unroll
        for (int t = 0; t < 4; ++t) {
          v16bf b = frag_b(f2w_l, 4 * TD, (wave * 4 + t) * 16,
                           half * 512 + k0, lane);
          facc[t] = wmma_bf(a, b, facc[t]);
        }
      }
      __syncthreads();
    }
    // residual add
    {
      const int c = lane & 15, mb = (lane >> 4) * 8;
#pragma unroll
      for (int t = 0; t < 4; ++t) {
        const int nb = (wave * 4 + t) * 16;
#pragma unroll
        for (int i = 0; i < 8; ++i) xs[mb + i][nb + c] += facc[t][i];
      }
    }
    __syncthreads();
  }

  // ---- store block -------------------------------------------------------
  {
    float4* op = (float4*)(out + blk * (TBS * TD));
    const float4* xp = (const float4*)(&xs[0][0]);
#pragma unroll
    for (int i = 0; i < (TBS * TD / 4) / NTHREADS; ++i)
      op[tid + i * NTHREADS] = xp[tid + i * NTHREADS];
  }
}

extern "C" void kernel_launch(void* const* d_in, const int* in_sizes, int n_in,
                              void* d_out, int out_size, void* d_ws, size_t ws_size,
                              hipStream_t stream) {
  const float* h    = (const float*)d_in[0];
  const float* n1w  = (const float*)d_in[1];
  const float* ipw_f = (const float*)d_in[2];
  const float* ipb  = (const float*)d_in[3];
  const float* opw_f = (const float*)d_in[4];
  const float* opb  = (const float*)d_in[5];
  const float* n2w  = (const float*)d_in[6];
  const float* f1w_f = (const float*)d_in[7];
  const float* f1b  = (const float*)d_in[8];
  const float* f2w_f = (const float*)d_in[9];
  const float* f2b  = (const float*)d_in[10];

  const int L    = in_sizes[1] / TD;              // 2
  const int nblk = in_sizes[0] / (TBS * TD);      // 4096

  // bf16 weight copies in workspace (~3 MB, lives in 192 MB L2 across blocks)
  bf16_t* wsb    = (bf16_t*)d_ws;
  bf16_t* ipw_bf = wsb;
  bf16_t* opw_bf = ipw_bf + (size_t)L * 3 * TD * TD;
  bf16_t* f1w_bf = opw_bf + (size_t)L * TD * TD;
  bf16_t* f2w_bf = f1w_bf + (size_t)L * 4 * TD * TD;

  const int n_ipw = in_sizes[2], n_opw = in_sizes[4];
  const int n_f1w = in_sizes[7], n_f2w = in_sizes[9];
  cvt_f32_bf16<<<(n_ipw + 255) / 256, 256, 0, stream>>>(ipw_f, ipw_bf, n_ipw);
  cvt_f32_bf16<<<(n_opw + 255) / 256, 256, 0, stream>>>(opw_f, opw_bf, n_opw);
  cvt_f32_bf16<<<(n_f1w + 255) / 256, 256, 0, stream>>>(f1w_f, f1w_bf, n_f1w);
  cvt_f32_bf16<<<(n_f2w + 255) / 256, 256, 0, stream>>>(f2w_f, f2w_bf, n_f2w);

  block_tf<<<nblk, NTHREADS, 0, stream>>>(h, n1w, ipb, opb, n2w, f1b, f2b,
                                          ipw_bf, opw_bf, f1w_bf, f2w_bf,
                                          (float*)d_out, L);
}
